// CIN_35450660061557
// MI455X (gfx1250) — matrix operand order repaired and checked by reading, not compiled, not run on hardware
//
#include <hip/hip_runtime.h>

typedef __attribute__((ext_vector_type(16))) __bf16 v16bf;
typedef __attribute__((ext_vector_type(8)))  __bf16 v8bf;
typedef __attribute__((ext_vector_type(8)))  float  v8f;

#define B_  512
#define M_  32
#define D_  64
#define O_  128
#define NT  256

// bf16 weight workspace layout (element offsets)
#define W0_OFF 0
#define W1_OFF (128 * 1024)
#define W2_OFF (W1_OFF + 128 * 4096)
#define WTOT   (W2_OFF + 128 * 4096)   // 1,179,648 bf16 = 2.25 MB (< ws_size)

// ---------------------------------------------------------------------------
// Pre-pass: convert all three fp32 weight matrices to bf16 in d_ws.
// Runs every kernel_launch (deterministic, graph-capture safe).
// ---------------------------------------------------------------------------
__global__ __launch_bounds__(256) void cin_cvt_w(
    const float* __restrict__ w0, const float* __restrict__ w1,
    const float* __restrict__ w2, unsigned short* __restrict__ dst)
{
  int i = blockIdx.x * 256 + threadIdx.x;
  float v;
  if (i < W1_OFF)      v = w0[i];
  else if (i < W2_OFF) v = w1[i - W1_OFF];
  else if (i < WTOT)   v = w2[i - W2_OFF];
  else return;
  __bf16 h = (__bf16)v;
  dst[i] = __builtin_bit_cast(unsigned short, h);
}

// ---------------------------------------------------------------------------
// Fused CIN: one workgroup per batch element, all 3 layers + final dot fused.
// out[o,d] = ReLU( sum_hh h[hh,d] * WMMA(W[:,hh,:], x0) + bias )  per layer.
// B operand (x0 tiles) is static in registers for the whole kernel.
// ---------------------------------------------------------------------------
__global__ __launch_bounds__(256) void cin_fused(
    const float* __restrict__ x0g,
    const float* __restrict__ bias0, const float* __restrict__ bias1,
    const float* __restrict__ bias2,
    const float* __restrict__ lw,   const float* __restrict__ lbp,
    const unsigned short* __restrict__ wbf_all,
    float* __restrict__ outp)
{
  __shared__ float hs[O_ * D_];   // 32 KB: layer i/o activations (starts as x0)
  __shared__ float wsum[8];

  const int b    = blockIdx.x;
  const int tid  = threadIdx.x;
  const int wave = tid >> 5;
  const int lane = tid & 31;
  const int lo   = lane & 15;     // column index within 16-wide tile
  const int kg   = lane >> 4;     // lane-group (k/row split per ISA layouts)

  // ---- x0 -> LDS (fp32, also serves as layer-0 h) ----
  const float* xb = x0g + (size_t)b * (M_ * D_);
  for (int i = tid; i < M_ * D_; i += NT) hs[i] = xb[i];
  __syncthreads();

  // ---- Static WMMA B operand: x0 tiles (32x16 KxN, bf16) ----
  // B layout: lanes 0-15 hold col N=lane, k={0..7,16..23}; lanes 16-31 col N=lane-16, k={8..15,24..31}
  v16bf bx[4];
  #pragma unroll
  for (int t = 0; t < 4; ++t) {
    const int d = t * 16 + lo;
    #pragma unroll
    for (int i = 0; i < 16; ++i) {
      const int m = kg * 8 + i + (i >= 8 ? 8 : 0);
      bx[t][i] = (__bf16)hs[m * D_ + d];
    }
  }

  const __bf16* wl[3] = {
      reinterpret_cast<const __bf16*>(wbf_all) + W0_OFF,
      reinterpret_cast<const __bf16*>(wbf_all) + W1_OFF,
      reinterpret_cast<const __bf16*>(wbf_all) + W2_OFF };
  const float* bl[3]    = { bias0, bias1, bias2 };
  const int    Hl[3]    = { 32, 128, 128 };
  const int    lwoff[3] = { 0, 128, 256 };

  float score = 0.f;

  for (int l = 0; l < 3; ++l) {
    const int H = Hl[l];
    const int K = H * M_;

    v8f acc[4];
    #pragma unroll
    for (int t = 0; t < 4; ++t)
      #pragma unroll
      for (int v = 0; v < 8; ++v) acc[t][v] = 0.f;

    // A operand row base: o = wave*16 + lo ; per-chunk k = hh*32 + kg*8 (+16)
    // (16-bit A layout: lanes 0-15 k={0..7,16..23}, lanes 16-31 k={8..15,24..31})
    const __bf16* wrow = wl[l] + (size_t)(wave * 16 + lo) * K + kg * 8;

    for (int hh = 0; hh < H; ++hh) {
      v8bf p0 = *reinterpret_cast<const v8bf*>(wrow + hh * M_);
      v8bf p1 = *reinterpret_cast<const v8bf*>(wrow + hh * M_ + 16);
      v16bf a = __builtin_shufflevector(p0, p1,
                  0,1,2,3,4,5,6,7,8,9,10,11,12,13,14,15);
      v8f zero = {};
      #pragma unroll
      for (int t = 0; t < 4; ++t) {
        v8f tile = __builtin_amdgcn_wmma_f32_16x16x32_bf16(
            false, a, false, bx[t], (short)0, zero, false, false);
        const float hsc = hs[hh * D_ + t * 16 + lo];   // f32 h scale (per column)
        #pragma unroll
        for (int v = 0; v < 8; ++v) acc[t][v] += hsc * tile[v];
      }
    }

    __syncthreads();   // all waves done reading hs(layer l)

    // bias + ReLU + fused final reduction (lw-weighted sum over d), write h_next
    const float* bias = bl[l];
    const float* lwl  = lw + lwoff[l];
    float part = 0.f;
    #pragma unroll
    for (int v = 0; v < 8; ++v) {
      const int   o  = wave * 16 + kg * 8 + v;   // C/D layout: M = v + 8*kg
      const float bo = bias[o];
      const float wo = lwl[o];
      #pragma unroll
      for (int t = 0; t < 4; ++t) {
        float val = acc[t][v] + bo;
        val = fmaxf(val, 0.f);
        part += wo * val;
        if (l < 2) hs[o * D_ + t * 16 + lo] = val;   // h for next layer
      }
    }
    score += part;
    __syncthreads();   // h_next fully written before next layer reads
  }

  // ---- deterministic block reduction of the score ----
  #pragma unroll
  for (int off = 16; off > 0; off >>= 1)
    score += __shfl_xor(score, off, 32);
  if (lane == 0) wsum[wave] = score;
  __syncthreads();
  if (tid == 0) {
    float s = lbp[0];
    #pragma unroll
    for (int w = 0; w < 8; ++w) s += wsum[w];
    outp[b] = s;
  }
}

// ---------------------------------------------------------------------------
extern "C" void kernel_launch(void* const* d_in, const int* in_sizes, int n_in,
                              void* d_out, int out_size, void* d_ws, size_t ws_size,
                              hipStream_t stream) {
  const float* x0 = (const float*)d_in[0];
  const float* W0 = (const float*)d_in[1];
  const float* b0 = (const float*)d_in[2];
  const float* W1 = (const float*)d_in[3];
  const float* b1 = (const float*)d_in[4];
  const float* W2 = (const float*)d_in[5];
  const float* b2 = (const float*)d_in[6];
  const float* lw = (const float*)d_in[7];
  const float* lb = (const float*)d_in[8];

  unsigned short* wbf = (unsigned short*)d_ws;   // 2.25 MB bf16 weights

  cin_cvt_w<<<(WTOT + 255) / 256, 256, 0, stream>>>(W0, W1, W2, wbf);
  cin_fused<<<B_, NT, 0, stream>>>(x0, b0, b1, b2, lw, lb, wbf, (float*)d_out);
}
